// GraphBased28x28x1_10170482557041
// MI455X (gfx1250) — compile-verified, matrix operand order checked by reading, not run on hardware
//
#include <hip/hip_runtime.h>
#include <hip/hip_bf16.h>

typedef __attribute__((ext_vector_type(16))) _Float16 v16h;
typedef __attribute__((ext_vector_type(8)))  float    v8f;

#define WMMA_F32_F16(a, b, c) \
  __builtin_amdgcn_wmma_f32_16x16x32_f16(false, (a), false, (b), (short)0, (c), false, false)

// Element e (0..15) of a 16-bit A/B fragment maps to K per ISA table
// (16-bit A-matrix 16x32): lanes 0-15: e<8 -> K=e, e>=8 -> K=8+e;
// lanes 16-31: e<8 -> K=8+e, e>=8 -> K=16+e.
__device__ __forceinline__ int frag_k(int lane, int e) {
  int base = (lane & 16) ? 8 : 0;
  return (e < 8) ? (base + e) : (base + 8 + e);
}
__device__ __forceinline__ float leakyf(float x) { return x >= 0.f ? x : 0.01f * x; }

// ---- CDNA5 async global->LDS copy (GV mode, 16B per lane, ASYNCcnt-tracked)
// VDST VGPR = LDS byte address (generic shared ptr low 32 bits == LDS offset,
// ISA 10.2 aperture rules); VADDR = 64-bit global address.
__device__ __forceinline__ void async_cp_b128(void* lds_ptr, const void* gptr) {
  uint32_t loff = (uint32_t)(uintptr_t)lds_ptr;
  uint64_t ga = (uint64_t)(uintptr_t)gptr;
  asm volatile("global_load_async_to_lds_b128 %0, %1, off" ::"v"(loff), "v"(ga)
               : "memory");
}
__device__ __forceinline__ void wait_async0() {
  asm volatile("s_wait_asynccnt 0" ::: "memory");
}

// ---------------------------------------------------------------- zero stats
__global__ void k_zero(float* p, int n) {
  int i = threadIdx.x + blockIdx.x * blockDim.x;
  if (i < n) p[i] = 0.f;
}

// ------------------------------------------------------- f32 -> f16 convert
__global__ void k_cvt(const float* __restrict__ a, _Float16* __restrict__ b, int n) {
  for (int i = threadIdx.x + blockIdx.x * blockDim.x; i < n; i += blockDim.x * gridDim.x)
    b[i] = (_Float16)a[i];
}

// -------------------------------------------- conv1 + relu + maxpool (WMMA)
// per image: [576 pix x 25 taps] * [25 x 20 ch], K pad 32, N pad 32
__global__ __launch_bounds__(128) void k_conv1(const _Float16* __restrict__ xh,
                                               const float* __restrict__ w,
                                               const float* __restrict__ bias,
                                               _Float16* __restrict__ h1) {
  __shared__ __align__(16) _Float16 simg[28 * 28];
  __shared__ __align__(16) _Float16 sconv[576 * 20];
  const int img = blockIdx.x, tid = threadIdx.x;
  const int lane = tid & 31, wave = tid >> 5;
  const _Float16* xi = xh + (size_t)img * 784;
  // 784 halves = 98 x 16B async chunks
  if (tid < 98) async_cp_b128(&simg[tid * 8], xi + tid * 8);
  wait_async0();
  __syncthreads();

  v16h bfrag[2];
#pragma unroll
  for (int nt = 0; nt < 2; ++nt) {
    int n = nt * 16 + (lane & 15);
#pragma unroll
    for (int e = 0; e < 16; ++e) {
      int k = frag_k(lane, e);
      float wv = (k < 25 && n < 20) ? w[n * 25 + k] : 0.f;
      bfrag[nt][e] = (_Float16)wv;
    }
  }

  for (int mt = wave; mt < 36; mt += 4) {  // 36 M-tiles, uniform per wave
    int m = mt * 16 + (lane & 15);
    int oy = m / 24, ox = m % 24;
    v16h a;
#pragma unroll
    for (int e = 0; e < 16; ++e) {
      int k = frag_k(lane, e);
      int dy = k / 5, dx = k % 5;
      a[e] = (k < 25) ? simg[(oy + dy) * 28 + (ox + dx)] : (_Float16)0.f;
    }
#pragma unroll
    for (int nt = 0; nt < 2; ++nt) {
      v8f c = {};
      c = WMMA_F32_F16(a, bfrag[nt], c);
      int n = nt * 16 + (lane & 15);
#pragma unroll
      for (int r = 0; r < 8; ++r) {
        int mm = mt * 16 + r + ((lane & 16) ? 8 : 0);
        if (n < 20) {
          float v = c[r] + bias[n];
          v = v > 0.f ? v : 0.f;
          sconv[mm * 20 + n] = (_Float16)v;
        }
      }
    }
  }
  __syncthreads();

  _Float16* ho = h1 + (size_t)img * 2880;  // [20][12][12]
  for (int o = tid; o < 2880; o += 128) {
    int c = o / 144, p = o % 144;
    int y = p / 12, xq = p % 12;
    float mx = -1e30f;
#pragma unroll
    for (int dy = 0; dy < 2; ++dy)
#pragma unroll
      for (int dx = 0; dx < 2; ++dx) {
        float v = (float)sconv[((2 * y + dy) * 24 + (2 * xq + dx)) * 20 + c];
        mx = v > mx ? v : mx;
      }
    ho[o] = (_Float16)mx;
  }
}

// -------------------------------------------- conv2 + relu + maxpool (WMMA)
// per image: [64 pix x 500] * [500 x 50], K pad 512 (16 steps), N pad 64
__global__ __launch_bounds__(128) void k_conv2(const _Float16* __restrict__ h1,
                                               const _Float16* __restrict__ w2h,
                                               const float* __restrict__ bias,
                                               _Float16* __restrict__ xflat) {
  __shared__ __align__(16) _Float16 sh1[2880];
  __shared__ __align__(16) _Float16 sw[50 * 500];
  __shared__ __align__(16) _Float16 sconv[64 * 50];
  const int img = blockIdx.x, tid = threadIdx.x;
  const int lane = tid & 31, wave = tid >> 5;
  const _Float16* hi = h1 + (size_t)img * 2880;
  // async stage: h1 tile 360 chunks, weights 3125 chunks (16B each)
  for (int c = tid; c < 360; c += 128) async_cp_b128(&sh1[c * 8], hi + c * 8);
  for (int c = tid; c < 3125; c += 128) async_cp_b128(&sw[c * 8], w2h + c * 8);
  wait_async0();
  __syncthreads();

  const int mt = wave;  // 4 M-tiles of 16 rows each
  const int m = mt * 16 + (lane & 15);
  const int oy = m >> 3, ox = m & 7;
  v8f acc[4] = {};
  for (int ks = 0; ks < 16; ++ks) {
    v16h a;
#pragma unroll
    for (int e = 0; e < 16; ++e) {
      int k = ks * 32 + frag_k(lane, e);
      _Float16 v = (_Float16)0.f;
      if (k < 500) {
        int ci = k / 25, t = k % 25;
        int dy = t / 5, dx = t % 5;
        v = sh1[ci * 144 + (oy + dy) * 12 + (ox + dx)];
      }
      a[e] = v;
    }
#pragma unroll
    for (int nt = 0; nt < 4; ++nt) {
      int n = nt * 16 + (lane & 15);
      v16h bf;
#pragma unroll
      for (int e = 0; e < 16; ++e) {
        int k = ks * 32 + frag_k(lane, e);
        bf[e] = (k < 500 && n < 50) ? sw[n * 500 + k] : (_Float16)0.f;
      }
      acc[nt] = WMMA_F32_F16(a, bf, acc[nt]);
    }
  }
#pragma unroll
  for (int nt = 0; nt < 4; ++nt) {
    int n = nt * 16 + (lane & 15);
#pragma unroll
    for (int r = 0; r < 8; ++r) {
      int mm = mt * 16 + r + ((lane & 16) ? 8 : 0);
      if (n < 50) {
        float v = acc[nt][r] + bias[n];
        v = v > 0.f ? v : 0.f;
        sconv[mm * 50 + n] = (_Float16)v;
      }
    }
  }
  __syncthreads();

  _Float16* xo = xflat + (size_t)img * 800;  // reshape(-1,800): c*16 + y*4 + x
  for (int o = tid; o < 800; o += 128) {
    int c = o / 16, p = o % 16;
    int y = p / 4, xq = p % 4;
    float mx = -1e30f;
#pragma unroll
    for (int dy = 0; dy < 2; ++dy)
#pragma unroll
      for (int dx = 0; dx < 2; ++dx) {
        float v = (float)sconv[((2 * y + dy) * 8 + (2 * xq + dx)) * 50 + c];
        mx = v > mx ? v : mx;
      }
    xo[c * 16 + p] = (_Float16)mx;
  }
}

// ------------------------------- fc: X = relu([4096x800]*[800x50]+b) (WMMA)
// also accumulates colsum(X) for the mean aggregation
__global__ __launch_bounds__(128) void k_fc(const _Float16* __restrict__ xflat,
                                            const _Float16* __restrict__ fch,
                                            const float* __restrict__ bias,
                                            _Float16* __restrict__ X,
                                            float* __restrict__ colsum) {
  const int tid = threadIdx.x, lane = tid & 31, wave = tid >> 5;
  const int row0 = blockIdx.x * 64 + wave * 16;
  const int mrow = row0 + (lane & 15);
  v8f acc[4] = {};
  for (int ks = 0; ks < 25; ++ks) {  // K = 800 = 25*32 exactly
    v16h a;
#pragma unroll
    for (int e = 0; e < 16; ++e) {
      int k = ks * 32 + frag_k(lane, e);
      a[e] = xflat[(size_t)mrow * 800 + k];
    }
#pragma unroll
    for (int nt = 0; nt < 4; ++nt) {
      int n = nt * 16 + (lane & 15);
      v16h bf;
#pragma unroll
      for (int e = 0; e < 16; ++e) {
        int k = ks * 32 + frag_k(lane, e);
        bf[e] = (n < 50) ? fch[n * 800 + k] : (_Float16)0.f;
      }
      acc[nt] = WMMA_F32_F16(a, bf, acc[nt]);
    }
  }
#pragma unroll
  for (int nt = 0; nt < 4; ++nt) {
    int n = nt * 16 + (lane & 15);
    float csum = 0.f;
#pragma unroll
    for (int r = 0; r < 8; ++r) {
      int mm = row0 + r + ((lane & 16) ? 8 : 0);
      float v = 0.f;
      if (n < 50) {
        v = acc[nt][r] + bias[n];
        v = v > 0.f ? v : 0.f;
      }
      X[(size_t)mm * 64 + n] = (_Float16)v;  // pad cols 50..63 with zero
      csum += v;
    }
    if (n < 50) atomicAdd(&colsum[n], csum);
  }
}

// ------- Z = leaky(X*root^T + (mean agg)*rel^T + b); softmax stats (WMMA)
__global__ __launch_bounds__(128) void k_z(const _Float16* __restrict__ X,
                                           const float* __restrict__ relw,
                                           const float* __restrict__ rootw,
                                           const float* __restrict__ rootb,
                                           const float* __restrict__ colsumX,
                                           float* __restrict__ zcol,
                                           float* __restrict__ u,
                                           float* __restrict__ G) {
  __shared__ float vbias[64];
  __shared__ float Zt[64 * 50];
  const int tid = threadIdx.x, lane = tid & 31, wave = tid >> 5;
  if (tid < 64) {
    float s = 0.f;
    if (tid < 50) {
      s = rootb[tid];
      for (int k = 0; k < 50; ++k)
        s += (colsumX[k] * (1.f / 4096.f)) * relw[tid * 50 + k];
    }
    vbias[tid] = s;
  }
  __syncthreads();

  const int row0 = blockIdx.x * 64 + wave * 16;
  const int mrow = row0 + (lane & 15);
  v8f acc[4] = {};
  for (int ks = 0; ks < 2; ++ks) {  // K = 50 padded to 64
    v16h a;
#pragma unroll
    for (int e = 0; e < 16; ++e) {
      int k = ks * 32 + frag_k(lane, e);
      a[e] = X[(size_t)mrow * 64 + k];  // zero-padded
    }
#pragma unroll
    for (int nt = 0; nt < 4; ++nt) {
      int n = nt * 16 + (lane & 15);
      v16h bf;
#pragma unroll
      for (int e = 0; e < 16; ++e) {
        int k = ks * 32 + frag_k(lane, e);
        bf[e] = (n < 50 && k < 50) ? (_Float16)rootw[n * 50 + k] : (_Float16)0.f;
      }
      acc[nt] = WMMA_F32_F16(a, bf, acc[nt]);
    }
  }
#pragma unroll
  for (int nt = 0; nt < 4; ++nt) {
    int n = nt * 16 + (lane & 15);
    if (n < 50) {
#pragma unroll
      for (int r = 0; r < 8; ++r) {
        int lr = wave * 16 + r + ((lane & 16) ? 8 : 0);
        Zt[lr * 50 + n] = leakyf(acc[nt][r] + vbias[n]);
      }
    }
  }
  __syncthreads();
  // Xp partial: colsum(Z)  (s = softmax over C=1 is all-ones)
  if (tid < 50) {
    float s = 0.f;
    for (int r = 0; r < 64; ++r) s += Zt[r * 50 + tid];
    atomicAdd(&zcol[tid], s);
  }
  __syncthreads();
  // in-place row softmax -> Ss
  if (tid < 64) {
    float* zr = &Zt[tid * 50];
    float m = zr[0];
    for (int j = 1; j < 50; ++j) m = zr[j] > m ? zr[j] : m;
    float s = 0.f;
    for (int j = 0; j < 50; ++j) { float e2 = expf(zr[j] - m); zr[j] = e2; s += e2; }
    float inv = 1.f / s;
    for (int j = 0; j < 50; ++j) zr[j] *= inv;
  }
  __syncthreads();
  if (tid < 50) {
    float s = 0.f;
    for (int r = 0; r < 64; ++r) s += Zt[r * 50 + tid];
    atomicAdd(&u[tid], s);
  }
  // G = Ss^T Ss partials (50x50)
  for (int o = tid; o < 2500; o += 128) {
    int i = o / 50, j = o % 50;
    float s = 0.f;
    for (int r = 0; r < 64; ++r) s += Zt[r * 50 + i] * Zt[r * 50 + j];
    atomicAdd(&G[o], s);
  }
}

// ---------------------------------------------------- single-block finisher
__global__ void k_final(const float* __restrict__ zcol, const float* __restrict__ u,
                        const float* __restrict__ G, const float* __restrict__ relw,
                        const float* __restrict__ rootw, const float* __restrict__ rootb,
                        const float* __restrict__ lin1w, const float* __restrict__ lin1b,
                        const float* __restrict__ lin2w, const float* __restrict__ lin2b,
                        float* __restrict__ out) {
  if (threadIdx.x != 0 || blockIdx.x != 0) return;
  const float n2 = 4096.f * 4096.f;
  float uu = 0.f;
  for (int k = 0; k < 50; ++k) uu += u[k] * u[k];
  float gg = 0.f;
  for (int k = 0; k < 2500; ++k) gg += G[k] * G[k];
  // aux_loss(ones, Z) = sqrt(N^2 - 2*||u||^2 + ||G||_F^2) / N^2 ; l1 = 0
  float loss1 = sqrtf(fmaxf(n2 - 2.f * uu + gg, 0.f)) / n2;

  float Xp[50];
  for (int k = 0; k < 50; ++k) Xp[k] = zcol[k];
  float X2[50];
  for (int j = 0; j < 50; ++j) {  // Ap@Xp/Ap = Xp -> combined rel+root
    float s = rootb[j];
    for (int k = 0; k < 50; ++k) s += Xp[k] * (relw[j * 50 + k] + rootw[j * 50 + k]);
    X2[j] = leakyf(s);
  }
  float m = X2[0];
  for (int j = 1; j < 50; ++j) m = fmaxf(m, X2[j]);
  float se = 0.f, ebuf[50];
  for (int j = 0; j < 50; ++j) { ebuf[j] = expf(X2[j] - m); se += ebuf[j]; }
  float ssq = 0.f;
  for (int j = 0; j < 50; ++j) { float p = ebuf[j] / se; ssq += p * p; }
  float loss2 = fabsf(n2 - ssq);  // aux_loss(Ap=N^2, X2), size 1

  float hb1[25];
  for (int i = 0; i < 25; ++i) {
    float s = lin1b[i];
    for (int j = 0; j < 50; ++j) s += X2[j] * lin1w[i * 50 + j];
    hb1[i] = leakyf(s);
  }
  float hb2[2];
  for (int k = 0; k < 2; ++k) {
    float s = lin2b[k];
    for (int i = 0; i < 25; ++i) s += hb1[i] * lin2w[k * 25 + i];
    hb2[k] = leakyf(s);
  }
  float mm = fmaxf(hb2[0], hb2[1]);
  float e0 = expf(hb2[0] - mm), e1 = expf(hb2[1] - mm), den = e0 + e1;
  out[0] = e0 / den;
  out[1] = e1 / den;
  out[2] = loss1 + loss2;  // l1 == 0
}

extern "C" void kernel_launch(void* const* d_in, const int* in_sizes, int n_in,
                              void* d_out, int out_size, void* d_ws, size_t ws_size,
                              hipStream_t stream) {
  const float* x       = (const float*)d_in[0];
  const float* conv1_w = (const float*)d_in[1];
  const float* conv1_b = (const float*)d_in[2];
  const float* conv2_w = (const float*)d_in[3];
  const float* conv2_b = (const float*)d_in[4];
  const float* fc_w    = (const float*)d_in[5];
  const float* fc_b    = (const float*)d_in[6];
  const float* rel_w   = (const float*)d_in[7];
  const float* root_w  = (const float*)d_in[8];
  const float* root_b  = (const float*)d_in[9];
  // d_in[10..14] (pool_*, mlp_*) provably do not affect outputs (softmax over C=1 -> ones)
  const float* lin1_w  = (const float*)d_in[15];
  const float* lin1_b  = (const float*)d_in[16];
  const float* lin2_w  = (const float*)d_in[17];
  const float* lin2_b  = (const float*)d_in[18];
  float* out = (float*)d_out;

  char* ws = (char*)d_ws;
  size_t off = 0;
  auto carve = [&](size_t bytes) {
    size_t o = off;
    off = (off + bytes + 255) & ~(size_t)255;
    return o;
  };
  _Float16* xh    = (_Float16*)(ws + carve((size_t)4096 * 784 * sizeof(_Float16)));
  _Float16* w2h   = (_Float16*)(ws + carve((size_t)25000 * sizeof(_Float16)));
  _Float16* fch   = (_Float16*)(ws + carve((size_t)40000 * sizeof(_Float16)));
  _Float16* h1    = (_Float16*)(ws + carve((size_t)4096 * 2880 * sizeof(_Float16)));
  _Float16* xflat = (_Float16*)(ws + carve((size_t)4096 * 800 * sizeof(_Float16)));
  _Float16* X     = (_Float16*)(ws + carve((size_t)4096 * 64 * sizeof(_Float16)));
  float* stats    = (float*)(ws + carve((size_t)2650 * sizeof(float)));
  float* colsumX = stats;        // 50
  float* zcol    = stats + 50;   // 50
  float* uvec    = stats + 100;  // 50
  float* G       = stats + 150;  // 2500

  k_zero<<<(2650 + 255) / 256, 256, 0, stream>>>(stats, 2650);
  k_cvt<<<512, 256, 0, stream>>>(x, xh, 4096 * 784);
  k_cvt<<<32, 256, 0, stream>>>(conv2_w, w2h, 25000);
  k_cvt<<<40, 256, 0, stream>>>(fc_w, fch, 40000);
  k_conv1<<<4096, 128, 0, stream>>>(xh, conv1_w, conv1_b, h1);
  k_conv2<<<4096, 128, 0, stream>>>(h1, w2h, conv2_b, xflat);
  k_fc<<<64, 128, 0, stream>>>(xflat, fch, fc_b, X, colsumX);
  k_z<<<64, 128, 0, stream>>>(X, rel_w, root_w, root_b, colsumX, zcol, uvec, G);
  k_final<<<1, 64, 0, stream>>>(zcol, uvec, G, rel_w, root_w, root_b,
                                lin1_w, lin1_b, lin2_w, lin2_b, out);
}
